// GIN_17257178595620
// MI455X (gfx1250) — matrix operand order repaired and checked by reading, not compiled, not run on hardware
//
#include <hip/hip_runtime.h>

typedef __attribute__((ext_vector_type(16))) __bf16 v16bf;
typedef __attribute__((ext_vector_type(8)))  __bf16 v8bf;
typedef __attribute__((ext_vector_type(8)))  float  v8f;

#define D_FEAT 128

__device__ __forceinline__ __bf16 f2bf(float f) {
    return static_cast<__bf16>(f);               // native bf16 cvt on gfx1250
}

// ---------------------------------------------------------------------------
// Weight conversion: f32 [L,K,N] -> bf16 transposed [L*3 mats][N][K]
// so each lane's 16-K B-fragment is one contiguous 32B load.
// ---------------------------------------------------------------------------
__global__ void convert_weights_kernel(const float* __restrict__ w1,
                                       const float* __restrict__ w2,
                                       const float* __restrict__ wres,
                                       __bf16* __restrict__ wt) {
    int gid = blockIdx.x * 256 + threadIdx.x;    // exactly 9*16384 threads
    int mat = gid >> 14;
    int r   = gid & 16383;
    int l   = mat / 3, m = mat % 3;
    int n   = r >> 7, k = r & 127;
    const float* src = (m == 0) ? w1 : (m == 1) ? w2 : wres;
    wt[gid] = f2bf(src[l * 16384 + k * 128 + n]);
}

__global__ void combine_bias_kernel(const float* __restrict__ b2,
                                    const float* __restrict__ rb,
                                    float* __restrict__ cb) {
    int i = blockIdx.x * 256 + threadIdx.x;
    if (i < 3 * D_FEAT) cb[i] = b2[i] + rb[i];
}

// ---------------------------------------------------------------------------
// h0 = x  (vectorized copy), then scatter: h0[dst] += x[src] over edges
// ---------------------------------------------------------------------------
__global__ void copy_kernel(const float* __restrict__ src, float* __restrict__ dst, int n4) {
    int i = blockIdx.x * blockDim.x + threadIdx.x;
    if (i < n4) ((float4*)dst)[i] = ((const float4*)src)[i];
}

__global__ void scatter_kernel(const float* __restrict__ x,
                               const long long* __restrict__ ei,
                               float* __restrict__ h0, int n_edges) {
    int gid = blockIdx.x * blockDim.x + threadIdx.x;
    int e = gid >> 5;                            // 32 threads per edge
    if (e >= n_edges) return;
    int q = (gid & 31) << 2;                     // 4 feats per thread
    long long s = ei[e];
    long long d = ei[(size_t)n_edges + e];
    float4 xv = *(const float4*)(x + (size_t)s * D_FEAT + q);
    float* hp = h0 + (size_t)d * D_FEAT + q;
    atomicAdd(hp + 0, xv.x);
    atomicAdd(hp + 1, xv.y);
    atomicAdd(hp + 2, xv.z);
    atomicAdd(hp + 3, xv.w);
}

// ---------------------------------------------------------------------------
// WMMA GEMM: out[N,128] = act( A1@W1t (+ A2@W2t) + bias )
// Block = 256 thr = 8 waves. Block tile: 32 rows x 128 cols.
// Wave w: row-tile rt = w>>2, col tiles at (w&3)*32 and (w&3)*32+16
// (two C tiles per wave sharing one A fragment -> 2 WMMAs per frag load).
// Full 32x128 K-tile of A staged once into LDS per pass -> barrier-free K loop.
// ---------------------------------------------------------------------------
__device__ __forceinline__ void gemm_pass(const float* __restrict__ A,
                                          const __bf16* __restrict__ Wt,
                                          int row0, int N, int tid,
                                          v8f& acc0, v8f& acc1) {
    __shared__ __bf16 ldsA[32 * D_FEAT];         // 8 KB
    const int lane = tid & 31;
    const int wave = tid >> 5;
    const int rt   = wave >> 2;                  // row tile 0/1
    const int r    = rt * 16 + (lane & 15);      // A row within block tile
    const int hi   = lane >> 4;
    const int nc0  = (wave & 3) * 32 + (lane & 15);
    const int nc1  = nc0 + 16;

    // protect LDS from previous pass, then stage full 32x128 A tile as bf16
    __syncthreads();
    {
        int ar = tid >> 3;                       // 0..31
        int ac = (tid & 7) * 16;                 // 16 K per thread
        int arow = row0 + ar;
        if (arow >= N) arow = N - 1;
        const float4* ap = (const float4*)(A + (size_t)arow * D_FEAT + ac);
        __bf16* dp = &ldsA[ar * D_FEAT + ac];
#pragma unroll
        for (int j = 0; j < 4; ++j) {
            float4 f = ap[j];
            dp[j * 4 + 0] = f2bf(f.x);
            dp[j * 4 + 1] = f2bf(f.y);
            dp[j * 4 + 2] = f2bf(f.z);
            dp[j * 4 + 3] = f2bf(f.w);
        }
    }
    __syncthreads();

#pragma unroll
    for (int kk = 0; kk < D_FEAT; kk += 32) {
        // A fragment: lane holds row r; K = kk + hi*16 .. +15
        v8bf alo = *reinterpret_cast<const v8bf*>(&ldsA[r * D_FEAT + kk + hi * 16]);
        v8bf ahi = *reinterpret_cast<const v8bf*>(&ldsA[r * D_FEAT + kk + hi * 16 + 8]);
        v16bf afrag;
#pragma unroll
        for (int i = 0; i < 8; ++i) { afrag[i] = alo[i]; afrag[i + 8] = ahi[i]; }

        // B fragments: lane holds cols nc0/nc1; K = kk + hi*16 .. +15 (32B each)
        v16bf b0 = *reinterpret_cast<const v16bf*>(Wt + (size_t)nc0 * D_FEAT + kk + hi * 16);
        v16bf b1 = *reinterpret_cast<const v16bf*>(Wt + (size_t)nc1 * D_FEAT + kk + hi * 16);

        acc0 = __builtin_amdgcn_wmma_f32_16x16x32_bf16(
            false, afrag, false, b0, (short)0, acc0, false, false);
        acc1 = __builtin_amdgcn_wmma_f32_16x16x32_bf16(
            false, afrag, false, b1, (short)0, acc1, false, false);
    }
}

__global__ void __launch_bounds__(256)
gemm_kernel(const float* __restrict__ A1, const __bf16* __restrict__ W1t,
            const float* __restrict__ A2, const __bf16* __restrict__ W2t,
            const float* __restrict__ bias, float* __restrict__ out,
            int N, int do_relu) {
    const int tid  = threadIdx.x;
    const int row0 = blockIdx.x * 32;
    v8f acc0 = {}, acc1 = {};

    gemm_pass(A1, W1t, row0, N, tid, acc0, acc1);
    if (A2) gemm_pass(A2, W2t, row0, N, tid, acc0, acc1);  // fused K=256

    const int lane = tid & 31;
    const int wave = tid >> 5;
    const int rt   = wave >> 2;
    const int hi   = lane >> 4;
    const int nc0  = (wave & 3) * 32 + (lane & 15);
    const int nc1  = nc0 + 16;
    const float b0 = bias[nc0];
    const float b1 = bias[nc1];
#pragma unroll
    for (int i = 0; i < 8; ++i) {
        int row = row0 + rt * 16 + i + hi * 8;   // C layout: lanes 0-15 M=v, 16-31 M=v+8
        float v0 = acc0[i] + b0;
        float v1 = acc1[i] + b1;
        if (do_relu) { v0 = fmaxf(v0, 0.0f); v1 = fmaxf(v1, 0.0f); }
        if (row < N) {
            out[(size_t)row * D_FEAT + nc0] = v0;
            out[(size_t)row * D_FEAT + nc1] = v1;
        }
    }
}

// ---------------------------------------------------------------------------
// Global mean pool + final linear
// ---------------------------------------------------------------------------
__global__ void zero_kernel(float* __restrict__ p, int n) {
    int i = blockIdx.x * blockDim.x + threadIdx.x;
    if (i < n) p[i] = 0.0f;
}

__global__ void pool_accum_kernel(const float* __restrict__ x,
                                  const long long* __restrict__ batch,
                                  float* __restrict__ pooled,
                                  float* __restrict__ counts, int n) {
    int gid = blockIdx.x * blockDim.x + threadIdx.x;
    int nid = gid >> 7;
    if (nid >= n) return;
    int f = gid & 127;
    int b = (int)batch[nid];
    atomicAdd(&pooled[b * D_FEAT + f], x[(size_t)nid * D_FEAT + f]);
    if (f == 0) atomicAdd(&counts[b], 1.0f);
}

__global__ void pool_out_kernel(const float* __restrict__ pooled,
                                const float* __restrict__ counts,
                                const float* __restrict__ out_w,
                                const float* __restrict__ out_b,
                                float* __restrict__ out) {
    int g = blockIdx.x;
    int f = threadIdx.x;                         // 128 threads
    __shared__ float s[D_FEAT];
    float denom = fmaxf(counts[g], 1.0f);
    s[f] = (pooled[g * D_FEAT + f] / denom) * out_w[f];
    __syncthreads();
    for (int off = 64; off > 0; off >>= 1) {
        if (f < off) s[f] += s[f + off];
        __syncthreads();
    }
    if (f == 0) out[g] = s[0] + out_b[0];
}

// ---------------------------------------------------------------------------
extern "C" void kernel_launch(void* const* d_in, const int* in_sizes, int n_in,
                              void* d_out, int out_size, void* d_ws, size_t ws_size,
                              hipStream_t stream) {
    const float*     x_in  = (const float*)d_in[0];
    const long long* ei    = (const long long*)d_in[1];
    const long long* batch = (const long long*)d_in[2];
    const float*     w1    = (const float*)d_in[3];
    const float*     b1    = (const float*)d_in[4];
    const float*     w2    = (const float*)d_in[5];
    const float*     b2    = (const float*)d_in[6];
    const float*     wres  = (const float*)d_in[7];
    const float*     rb    = (const float*)d_in[8];
    const float*     out_w = (const float*)d_in[9];
    const float*     out_b = (const float*)d_in[10];

    const int n       = in_sizes[0] / D_FEAT;     // 50000
    const int n_edges = in_sizes[1] / 2;          // 600000
    const int n_graph = out_size;                 // 64

    // workspace layout
    char* ws = (char*)d_ws;
    size_t nodeBytes = (((size_t)n * D_FEAT * 4) + 255) & ~(size_t)255;
    float*  buf0 = (float*)(ws);
    float*  buf1 = (float*)(ws + nodeBytes);
    float*  buf2 = (float*)(ws + 2 * nodeBytes);
    size_t  o    = 3 * nodeBytes;
    __bf16* wt   = (__bf16*)(ws + o); o += (size_t)9 * 16384 * 2;   // 9 bf16 mats
    float*  cb   = (float*)(ws + o);  o += (size_t)3 * D_FEAT * 4;  // b2 + res_b
    float*  pooled = (float*)(ws + o); o += (size_t)n_graph * D_FEAT * 4;
    float*  counts = (float*)(ws + o);

    convert_weights_kernel<<<(9 * 16384) / 256, 256, 0, stream>>>(w1, w2, wres, wt);
    combine_bias_kernel<<<2, 256, 0, stream>>>(b2, rb, cb);

    // buffer rotation per layer (xcur never written)
    const float* XC[3] = { x_in, buf2, buf0 };
    float*       H0[3] = { buf0, buf0, buf2 };
    float*       XN[3] = { buf2, buf0, buf2 };

    const int gemmGrid  = (n + 31) / 32;
    const int copyGrid  = (n * 32 + 255) / 256;          // n*128/4 float4s
    const int scatGrid  = (n_edges * 32 + 255) / 256;    // 32 threads / edge

    for (int l = 0; l < 3; ++l) {
        copy_kernel<<<copyGrid, 256, 0, stream>>>(XC[l], H0[l], n * 32);
        scatter_kernel<<<scatGrid, 256, 0, stream>>>(XC[l], ei, H0[l], n_edges);
        // t = relu(h0 @ W1 + b1)
        gemm_kernel<<<gemmGrid, 256, 0, stream>>>(
            H0[l], wt + (size_t)(l * 3 + 0) * 16384, nullptr, nullptr,
            b1 + l * D_FEAT, buf1, n, 1);
        // x' = act( t @ W2 + x @ Wres + (b2 + res_b) )
        gemm_kernel<<<gemmGrid, 256, 0, stream>>>(
            buf1, wt + (size_t)(l * 3 + 1) * 16384,
            XC[l], wt + (size_t)(l * 3 + 2) * 16384,
            cb + l * D_FEAT, XN[l], n, (l < 2) ? 1 : 0);
    }

    zero_kernel<<<(n_graph * D_FEAT + n_graph + 255) / 256, 256, 0, stream>>>(
        pooled, n_graph * D_FEAT + n_graph);             // pooled + counts contiguous
    pool_accum_kernel<<<((size_t)n * D_FEAT + 255) / 256, 256, 0, stream>>>(
        XN[2], batch, pooled, counts, n);
    pool_out_kernel<<<n_graph, D_FEAT, 0, stream>>>(pooled, counts, out_w, out_b,
                                                    (float*)d_out);
}
